// QLinear_35562329211254
// MI455X (gfx1250) — matrix operand order repaired and checked by reading, not compiled, not run on hardware
//
#include <hip/hip_runtime.h>
#include <stdint.h>

typedef __attribute__((ext_vector_type(16))) __bf16 v16bf;
typedef __attribute__((ext_vector_type(8)))  __bf16 v8bf;
typedef __attribute__((ext_vector_type(8)))  float  v8f;
typedef __attribute__((ext_vector_type(4)))  float  v4f;
typedef __attribute__((ext_vector_type(4)))  int    v4i;

#define TOKENS   8192
#define OUT_F    4096
#define IN_F     4096
#define GHALF    131072   // G/2 = (OUT_F*IN_F/64)/2
#define BM       128
#define BN       128
#define BK       64
#define KTILES   (IN_F / BK)   // 64
#define LDSW     72            // 64 + 8 pad (row stride = 144B = 9*16B, keeps 16B alignment)

__device__ __forceinline__ __bf16 f2bf(float f) {
  uint32_t u = __builtin_bit_cast(uint32_t, f);
  uint32_t r = u + 0x7FFFu + ((u >> 16) & 1u);   // round-to-nearest-even
  return __builtin_bit_cast(__bf16, (uint16_t)(r >> 16));
}

__global__ __launch_bounds__(256)
void hqq_gemm_bf16_wmma(const float* __restrict__ x,
                        const int*   __restrict__ Wq,
                        const float* __restrict__ scale,
                        const float* __restrict__ zero,
                        const float* __restrict__ bias,
                        float* __restrict__ out) {
  __shared__ __bf16 As[BM][LDSW];   // activations, bf16
  __shared__ __bf16 Bs[BN][LDSW];   // dequantized weights, bf16 (row = out-feature, col = k)

  const int tid   = threadIdx.x;
  const int lane  = tid & 31;
  const int wave  = tid >> 5;
  const int mwave = wave >> 2;       // 0..1  -> 64-row M slab
  const int nwave = wave & 3;        // 0..3  -> 32-col N slab
  const int laneN = lane & 15;       // column / row-in-tile selector
  const int laneH = lane >> 4;       // half selector (A: K-half of 8, B: K-half of 8, C: M-half of 8)

  const int bm = blockIdx.y * BM;    // token base
  const int bn = blockIdx.x * BN;    // out-feature base

  // ---- A staging map: 8192 floats/tile, 8 x float4 per thread ----
  const int rowA = tid >> 4;             // 0..15 (step 16 per iter)
  const int colA = (tid & 15) * 4;       // 0..60

  // ---- B staging map: 2 threads per out-row, 32 packed ints each ----
  const int rowB  = tid >> 1;            // 0..127
  const int colB0 = (tid & 1) * 32;      // 0 or 32
  const int oB    = bn + rowB;           // absolute out-feature of this stager

  v8f acc[4][2];
  {
    v8f zf = {0.f,0.f,0.f,0.f,0.f,0.f,0.f,0.f};
    #pragma unroll
    for (int mt = 0; mt < 4; ++mt)
      #pragma unroll
      for (int nt = 0; nt < 2; ++nt)
        acc[mt][nt] = zf;
  }

  const int mBaseW = mwave * 64;
  const int nBaseW = nwave * 32;

  for (int kt = 0; kt < KTILES; ++kt) {
    // ---------- stage A: x fp32 -> bf16 ----------
    #pragma unroll
    for (int t = 0; t < 8; ++t) {
      const int row = rowA + t * 16;
      const v4f xv = *(const v4f*)&x[(size_t)(bm + row) * IN_F + kt * BK + colA];
      As[row][colA + 0] = f2bf(xv.x);
      As[row][colA + 1] = f2bf(xv.y);
      As[row][colA + 2] = f2bf(xv.z);
      As[row][colA + 3] = f2bf(xv.w);
    }

    // ---------- stage B: unpack 4-bit, dequant -> bf16 ----------
    {
      const int g  = oB * 64 + kt;                    // group id (axis-1 grouping)
      const int r  = (g < GHALF) ? g : (g - GHALF);   // packed row
      const int sh = (g < GHALF) ? 4 : 0;             // hi nibble = first half of groups
      const float sc = scale[g];
      const float zp = zero[g];
      const v4i* wp = (const v4i*)&Wq[(size_t)r * 64 + colB0];
      #pragma unroll
      for (int t = 0; t < 8; ++t) {
        const v4i q = wp[t];
        const int c = colB0 + t * 4;
        Bs[rowB][c + 0] = f2bf(((float)((q.x >> sh) & 0xF) - zp) * sc);
        Bs[rowB][c + 1] = f2bf(((float)((q.y >> sh) & 0xF) - zp) * sc);
        Bs[rowB][c + 2] = f2bf(((float)((q.z >> sh) & 0xF) - zp) * sc);
        Bs[rowB][c + 3] = f2bf(((float)((q.w >> sh) & 0xF) - zp) * sc);
      }
    }

    // ---------- prefetch next K-tile while this one is consumed ----------
    if (kt + 1 < KTILES) {
      __builtin_prefetch(&x[(size_t)(bm + rowA) * IN_F + (kt + 1) * BK + colA], 0, 1);
      const int gn = oB * 64 + kt + 1;
      const int rn = (gn < GHALF) ? gn : (gn - GHALF);
      __builtin_prefetch(&Wq[(size_t)rn * 64 + colB0], 0, 1);
    }

    __syncthreads();

    // ---------- WMMA: 2 K-steps of 32, 4x2 tiles of 16x16 ----------
    #pragma unroll
    for (int k0 = 0; k0 < BK; k0 += 32) {
      v16bf a[4];
      #pragma unroll
      for (int mt = 0; mt < 4; ++mt) {
        const int m = mBaseW + mt * 16 + laneN;
        // A layout: VGPR0-3 hold K = 8*half + 0..7 ; VGPR4-7 hold K = 16 + 8*half + 0..7
        const v8bf lo = *(const v8bf*)&As[m][k0 + laneH * 8];
        const v8bf hi = *(const v8bf*)&As[m][k0 + 16 + laneH * 8];
        a[mt] = __builtin_shufflevector(lo, hi, 0,1,2,3,4,5,6,7,8,9,10,11,12,13,14,15);
      }
      #pragma unroll
      for (int nt = 0; nt < 2; ++nt) {
        const int n = nBaseW + nt * 16 + laneN;
        // B layout: VGPR0-3 hold K = 8*half + 0..7 ; VGPR4-7 hold K = 16 + 8*half + 0..7
        const v8bf b0 = *(const v8bf*)&Bs[n][k0 + laneH * 8];
        const v8bf b1 = *(const v8bf*)&Bs[n][k0 + 16 + laneH * 8];
        const v16bf b = __builtin_shufflevector(b0, b1, 0,1,2,3,4,5,6,7,8,9,10,11,12,13,14,15);
        #pragma unroll
        for (int mt = 0; mt < 4; ++mt) {
          acc[mt][nt] = __builtin_amdgcn_wmma_f32_16x16x32_bf16(
              /*neg_a=*/false, a[mt], /*neg_b=*/false, b,
              /*c_mod=*/(short)0, acc[mt][nt],
              /*reuse_a=*/false, /*reuse_b=*/false);
        }
      }
    }

    __syncthreads();
  }

  // ---------- epilogue: bias add, fp32 store ----------
  #pragma unroll
  for (int nt = 0; nt < 2; ++nt) {
    const int o  = bn + nBaseW + nt * 16 + laneN;
    const float bv = bias[o];
    #pragma unroll
    for (int mt = 0; mt < 4; ++mt) {
      const int m0 = bm + mBaseW + mt * 16 + laneH * 8;  // C layout: VGPR j -> row 8*half + j
      #pragma unroll
      for (int j = 0; j < 8; ++j) {
        out[(size_t)(m0 + j) * OUT_F + o] = acc[mt][nt][j] + bv;
      }
    }
  }
}

extern "C" void kernel_launch(void* const* d_in, const int* in_sizes, int n_in,
                              void* d_out, int out_size, void* d_ws, size_t ws_size,
                              hipStream_t stream) {
  const float* x     = (const float*)d_in[0];
  const int*   Wq    = (const int*)  d_in[1];
  const float* scale = (const float*)d_in[2];
  const float* zr    = (const float*)d_in[3];
  const float* bias  = (const float*)d_in[4];
  float* out = (float*)d_out;

  dim3 grid(OUT_F / BN, TOKENS / BM);   // (32, 64)
  dim3 block(256);
  hipLaunchKernelGGL(hqq_gemm_bf16_wmma, grid, block, 0, stream,
                     x, Wq, scale, zr, bias, out);
}
